// GIN_Net_87101936763026
// MI455X (gfx1250) — compile-verified, matrix operand order checked by reading, not run on hardware
//
#include <hip/hip_runtime.h>

#define GIN_N 100000
#define GIN_E 1600000
#define GIN_H 64
#define GIN_BN_EPS 1e-5f

typedef __attribute__((ext_vector_type(16))) __bf16 v16bf;
typedef __attribute__((ext_vector_type(8)))  float  v8f;

// ---------------------------------------------------------------------------
// agg <- feat  (vectorized float4 copy; initializes the scatter destination)
// ---------------------------------------------------------------------------
__global__ __launch_bounds__(256) void copy_feat_kernel(const float* __restrict__ src,
                                                        float* __restrict__ dst, int n4) {
  int i = blockIdx.x * 256 + threadIdx.x;
  if (i < n4) ((float4*)dst)[i] = ((const float4*)src)[i];
}

// ---------------------------------------------------------------------------
// agg[dst[e]] += feat[src[e]]   (64 features/edge, 16 threads/edge x float4)
// agg (25.6 MB) is L2-resident on MI455X (192 MB L2) -> L2 atomic path.
// ---------------------------------------------------------------------------
__global__ __launch_bounds__(256) void scatter_add_kernel(const float* __restrict__ feat,
                                                          const int* __restrict__ srcv,
                                                          const int* __restrict__ dstv,
                                                          float* __restrict__ agg, int E) {
  unsigned tid = blockIdx.x * 256u + threadIdx.x;
  unsigned e   = tid >> 4;
  if (e >= (unsigned)E) return;
  int f = (int)(tid & 15u) * 4;
  int s = srcv[e];
  int d = dstv[e];
  float4 v = *(const float4*)(feat + (size_t)s * GIN_H + f);
  float* p = agg + (size_t)d * GIN_H + f;
  atomicAdd(p + 0, v.x);
  atomicAdd(p + 1, v.y);
  atomicAdd(p + 2, v.z);
  atomicAdd(p + 3, v.w);
}

// ---------------------------------------------------------------------------
// Pre-pack a (64 x nout) f32 weight matrix into per-lane WMMA B fragments:
//   packed[(ct*2 + t)*32 + lane] = v16bf B-fragment for col-tile ct, K-tile t.
// One thread per (tile, lane); 32-byte contiguous store per thread.
// ---------------------------------------------------------------------------
__global__ void pack_w_kernel(const float* __restrict__ W, int nout, int ntiles,
                              __bf16* __restrict__ packed) {
  int tid = blockIdx.x * blockDim.x + threadIdx.x;  // tile*32 + lane
  if (tid >= ntiles * 32) return;
  int lane = tid & 31;
  int tile = tid >> 5;
  int t    = tile & 1;     // K-tile (0: K=0..31, 1: K=32..63)
  int ct   = tile >> 1;    // column tile
  int row  = lane & 15;
  int kb   = (lane >> 4) * 8;
  int cn   = ct * 16 + row;
  int k0   = t * 32;
  bool cok = cn < nout;
  v16bf o;
#pragma unroll
  for (int j = 0; j < 8; ++j) {
    o[j]     = cok ? (__bf16)W[(k0 + kb + j) * nout + cn]      : (__bf16)0.f;
    o[8 + j] = cok ? (__bf16)W[(k0 + kb + 16 + j) * nout + cn] : (__bf16)0.f;
  }
  ((v16bf*)packed)[tid] = o;
}

// ---------------------------------------------------------------------------
// Fold bias + BatchNorm into per-column scale/shift: scsh[c]=sc, scsh[64+c]=sh
// ---------------------------------------------------------------------------
__global__ void pack_bn_kernel(const float* __restrict__ b1, const float* __restrict__ g,
                               const float* __restrict__ bt, const float* __restrict__ m,
                               const float* __restrict__ v, float* __restrict__ scsh) {
  int c = threadIdx.x;
  if (c < GIN_H) {
    float sc = g[c] * rsqrtf(v[c] + GIN_BN_EPS);
    scsh[c]         = sc;
    scsh[GIN_H + c] = (b1[c] - m[c]) * sc + bt[c];
  }
}

// ---------------------------------------------------------------------------
// WMMA helper: D = A(16x32 bf16) * B(32x16 bf16) + C(16x16 f32)
// ---------------------------------------------------------------------------
__device__ __forceinline__ v8f wmma_bf16(v16bf a, v16bf b, v8f c) {
  return __builtin_amdgcn_wmma_f32_16x16x32_bf16(false, a, false, b, (short)0, c,
                                                 false, false);
}

// ---------------------------------------------------------------------------
// Fused GIN MLP:  out = [ReLU]( ReLU(BN(agg @ w1 + b1)) @ w2 + b2 )
// One wave per 16-node tile; 8 waves (128 nodes) per 256-thread block.
// Weights arrive pre-packed in WMMA B-fragment layout (bf16).
// ---------------------------------------------------------------------------
template <int NOUT, bool RELU_OUT>
__global__ __launch_bounds__(256) void gin_mlp_kernel(
    const float* __restrict__ agg,
    const v16bf* __restrict__ w1p,    // 8 tiles  (4 ct x 2 kt)
    const float* __restrict__ scsh,   // [64] scale, [64] shift
    const v16bf* __restrict__ w2p,    // NCT*2 tiles
    const float* __restrict__ b2,
    float* __restrict__ out) {
  __shared__ __bf16 zlds[8][16][GIN_H + 8];  // hidden activations, bf16, padded

  const int wave = threadIdx.x >> 5;
  const int lane = threadIdx.x & 31;
  const int row  = lane & 15;        // M (for A/C) or N (for B/C) within tile
  const int kb   = (lane >> 4) * 8;  // K-half base (0 or 8); also C row base

  const int  node0  = blockIdx.x * 128 + wave * 16;
  const bool active = (node0 < GIN_N);          // wave-uniform
  const int  nb     = active ? node0 : 0;       // keep waves convergent

  // ---- load A tiles (K = 0..31 and 32..63) in ISA 16-bit A layout ----
  const float* arow = agg + (size_t)(nb + row) * GIN_H;
  v16bf a0, a1;
#pragma unroll
  for (int j = 0; j < 8; ++j) {
    a0[j]     = (__bf16)arow[kb + j];
    a0[8 + j] = (__bf16)arow[kb + 16 + j];
    a1[j]     = (__bf16)arow[32 + kb + j];
    a1[8 + j] = (__bf16)arow[32 + kb + 16 + j];
  }

  // ---- GEMM1 (K=64) + fused bias/BN/ReLU -> LDS (bf16) ----
#pragma unroll
  for (int ct = 0; ct < 4; ++ct) {
    const int cn = ct * 16 + row;  // output column 0..63
    v16bf bt0 = w1p[(ct * 2 + 0) * 32 + lane];
    v16bf bt1 = w1p[(ct * 2 + 1) * 32 + lane];
    v8f c = {0.f, 0.f, 0.f, 0.f, 0.f, 0.f, 0.f, 0.f};
    c = wmma_bf16(a0, bt0, c);
    c = wmma_bf16(a1, bt1, c);
    const float sc = scsh[cn];
    const float sh = scsh[GIN_H + cn];
#pragma unroll
    for (int i = 0; i < 8; ++i) {
      float z = c[i] * sc + sh;
      z = z > 0.f ? z : 0.f;
      zlds[wave][kb + i][cn] = (__bf16)z;  // row M = kb+i, col = cn
    }
  }

  __syncthreads();

  // ---- re-lay hidden activations into A format from LDS ----
  const __bf16* zrow = &zlds[wave][row][0];
  v16bf h0, h1;
#pragma unroll
  for (int j = 0; j < 8; ++j) {
    h0[j]     = zrow[kb + j];
    h0[8 + j] = zrow[kb + 16 + j];
    h1[j]     = zrow[32 + kb + j];
    h1[8 + j] = zrow[32 + kb + 16 + j];
  }

  // ---- GEMM2 (K=64, NOUT cols) + bias (+ReLU) -> global ----
  constexpr int NCT = (NOUT + 15) / 16;
#pragma unroll
  for (int ct = 0; ct < NCT; ++ct) {
    const int  cn  = ct * 16 + row;
    const bool cok = (cn < NOUT);
    v16bf bt0 = w2p[(ct * 2 + 0) * 32 + lane];
    v16bf bt1 = w2p[(ct * 2 + 1) * 32 + lane];
    v8f c = {0.f, 0.f, 0.f, 0.f, 0.f, 0.f, 0.f, 0.f};
    c = wmma_bf16(h0, bt0, c);
    c = wmma_bf16(h1, bt1, c);
    if (active && cok) {
      const float bb = b2[cn];
#pragma unroll
      for (int i = 0; i < 8; ++i) {
        float z = c[i] + bb;
        if (RELU_OUT) z = z > 0.f ? z : 0.f;
        out[(size_t)(node0 + kb + i) * NOUT + cn] = z;
      }
    }
  }
}

// ---------------------------------------------------------------------------
// In-place log_softmax over C columns, one thread per row.
// ---------------------------------------------------------------------------
template <int C>
__global__ __launch_bounds__(256) void log_softmax_kernel(float* __restrict__ out, int n) {
  int i = blockIdx.x * 256 + threadIdx.x;
  if (i >= n) return;
  float* r = out + (size_t)i * C;
  float buf[C];
  float mx = -3.402823466e38f;
#pragma unroll
  for (int c = 0; c < C; ++c) {
    buf[c] = r[c];
    mx = fmaxf(mx, buf[c]);
  }
  float s = 0.f;
#pragma unroll
  for (int c = 0; c < C; ++c) s += __expf(buf[c] - mx);
  const float lse = __logf(s) + mx;
#pragma unroll
  for (int c = 0; c < C; ++c) r[c] = buf[c] - lse;
}

// ---------------------------------------------------------------------------
extern "C" void kernel_launch(void* const* d_in, const int* in_sizes, int n_in,
                              void* d_out, int out_size, void* d_ws, size_t ws_size,
                              hipStream_t stream) {
  const float* x    = (const float*)d_in[0];
  const int*   ei   = (const int*)d_in[1];
  const int*   srcv = ei;
  const int*   dstv = ei + GIN_E;
  const float* w1_0 = (const float*)d_in[2];
  const float* b1_0 = (const float*)d_in[3];
  const float* g_0  = (const float*)d_in[4];
  const float* bb_0 = (const float*)d_in[5];
  const float* m_0  = (const float*)d_in[6];
  const float* v_0  = (const float*)d_in[7];
  const float* w2_0 = (const float*)d_in[8];
  const float* b2_0 = (const float*)d_in[9];
  const float* w1_1 = (const float*)d_in[10];
  const float* b1_1 = (const float*)d_in[11];
  const float* g_1  = (const float*)d_in[12];
  const float* bb_1 = (const float*)d_in[13];
  const float* m_1  = (const float*)d_in[14];
  const float* v_1  = (const float*)d_in[15];
  const float* w2_1 = (const float*)d_in[16];
  const float* b2_1 = (const float*)d_in[17];

  // ---- workspace layout (d_ws is >=256B aligned; all offsets 32B aligned) ----
  float*  agg  = (float*)d_ws;                          // N*64 f32 (25.6 MB)
  float*  hbuf = agg + (size_t)GIN_N * GIN_H;           // N*64 f32 (25.6 MB)
  __bf16* w1p0 = (__bf16*)(hbuf + (size_t)GIN_N * GIN_H);
  __bf16* w2p0 = w1p0 + 8 * 32 * 16;                    // 4096 bf16 each
  __bf16* w1p1 = w2p0 + 8 * 32 * 16;
  __bf16* w2p1 = w1p1 + 8 * 32 * 16;                    // 6 tiles used, pad to 8
  float*  scsh0 = (float*)(w2p1 + 8 * 32 * 16);         // 128 f32
  float*  scsh1 = scsh0 + 2 * GIN_H;                    // 128 f32

  const int n4        = GIN_N * GIN_H / 4;
  const int cpBlocks  = (n4 + 255) / 256;
  const int scBlocks  = (GIN_E * 16) / 256;             // exact: 100000
  const int mlpBlocks = (GIN_N + 127) / 128;
  const int smBlocks  = (GIN_N + 255) / 256;

  // ---- one-time (per launch) weight/BN packing: 6 tiny kernels ----
  pack_w_kernel<<<1, 256, 0, stream>>>(w1_0, GIN_H, 8, w1p0);
  pack_w_kernel<<<1, 256, 0, stream>>>(w2_0, GIN_H, 8, w2p0);
  pack_w_kernel<<<1, 256, 0, stream>>>(w1_1, GIN_H, 8, w1p1);
  pack_w_kernel<<<1, 256, 0, stream>>>(w2_1, 40,    6, w2p1);
  pack_bn_kernel<<<1, GIN_H, 0, stream>>>(b1_0, g_0, bb_0, m_0, v_0, scsh0);
  pack_bn_kernel<<<1, GIN_H, 0, stream>>>(b1_1, g_1, bb_1, m_1, v_1, scsh1);

  // ---- layer 0 ----
  copy_feat_kernel<<<cpBlocks, 256, 0, stream>>>(x, agg, n4);  // agg = x (eps=0 self term)
  scatter_add_kernel<<<scBlocks, 256, 0, stream>>>(x, srcv, dstv, agg, GIN_E);
  gin_mlp_kernel<GIN_H, true><<<mlpBlocks, 256, 0, stream>>>(
      agg, (const v16bf*)w1p0, scsh0, (const v16bf*)w2p0, b2_0, hbuf);

  // ---- layer 1 ----
  copy_feat_kernel<<<cpBlocks, 256, 0, stream>>>(hbuf, agg, n4);
  scatter_add_kernel<<<scBlocks, 256, 0, stream>>>(hbuf, srcv, dstv, agg, GIN_E);
  gin_mlp_kernel<40, false><<<mlpBlocks, 256, 0, stream>>>(
      agg, (const v16bf*)w1p1, scsh1, (const v16bf*)w2p1, b2_1, (float*)d_out);

  // ---- log_softmax ----
  log_softmax_kernel<40><<<smBlocks, 256, 0, stream>>>((float*)d_out, GIN_N);
}
